// MyGRU_58463094833755
// MI455X (gfx1250) — matrix operand order repaired and checked by reading, not compiled, not run on hardware
//
#include <hip/hip_runtime.h>

// ---------------------------------------------------------------------------
// GRU on gfx1250 (MI455X): bf16 WMMA (v_wmma_f32_16x16x32_bf16) everywhere.
//   Kernel A: X -> bf16 copy (one pass; 64MB, L2-resident afterwards)
//   Kernel 1: xr/xu/xh = X @ Wx{r,u,h} + b  (fused 3-gate tiled GEMM,
//             A-panel staged with global_load_async_to_lds_b128 / ASYNCcnt)
//   Kernel 2: persistent scan, recurrent weights resident in LDS,
//             software global barrier between the two GEMM phases per step.
// ---------------------------------------------------------------------------

#define T_ 1024
#define B_ 64
#define I_ 512
#define H_ 512

typedef __attribute__((ext_vector_type(16))) __bf16 v16bf;
typedef __attribute__((ext_vector_type(8)))  float  v8f;

union FragAB {
    v16bf v;
    uint4 q[2];  // two 16-byte chunks
};

__device__ __forceinline__ unsigned short f2bf(float f) {
    unsigned int u = __float_as_uint(f);
    // round-to-nearest-even to bf16
    unsigned int r = (u + 0x7FFFu + ((u >> 16) & 1u)) >> 16;
    return (unsigned short)r;
}

__device__ __forceinline__ float sigmoidf_(float x) {
    return 1.0f / (1.0f + __expf(-x));
}

// Low 32 bits of a generic pointer to __shared__ = LDS byte offset
// (LDS aperture maps addr[31:0] directly; ISA 10.2 aperture table).
__device__ __forceinline__ unsigned lds_lo32(const void* p) {
    return (unsigned)(unsigned long long)p;
}

// Async copy 16B global -> LDS (per-lane addresses), tracked by ASYNCcnt.
__device__ __forceinline__ void async_ld_b128(unsigned lds_off,
                                              const void* gaddr) {
    asm volatile("global_load_async_to_lds_b128 %0, %1, off"
                 :: "v"(lds_off), "v"(gaddr)
                 : "memory");
}

__device__ __forceinline__ void wait_async0() {
    asm volatile("s_wait_asynccnt 0x0" ::: "memory");
}

// ---------------------------------------------------------------------------
// Kernel 0: init bf16 state from initial_state, zero barrier counter
// ---------------------------------------------------------------------------
__global__ void gru_init(const float* __restrict__ s0,
                         unsigned short* __restrict__ state_bf,
                         unsigned int* __restrict__ counter) {
    int idx = blockIdx.x * blockDim.x + threadIdx.x;
    if (idx < B_ * H_) state_bf[idx] = f2bf(s0[idx]);
    if (idx == 0) *counter = 0u;
}

// ---------------------------------------------------------------------------
// Kernel A: convert X [T*B, I] fp32 -> bf16 (8 elements / thread)
// ---------------------------------------------------------------------------
__global__ void x_to_bf16(const float* __restrict__ X,
                          unsigned short* __restrict__ Xb) {
    size_t idx = ((size_t)blockIdx.x * blockDim.x + threadIdx.x) * 8;
    float4 a = *(const float4*)(X + idx);
    float4 b = *(const float4*)(X + idx + 4);
    union { unsigned short s[8]; uint4 q; } o;
    o.s[0] = f2bf(a.x); o.s[1] = f2bf(a.y);
    o.s[2] = f2bf(a.z); o.s[3] = f2bf(a.w);
    o.s[4] = f2bf(b.x); o.s[5] = f2bf(b.y);
    o.s[6] = f2bf(b.z); o.s[7] = f2bf(b.w);
    *(uint4*)(Xb + idx) = o.q;
}

// ---------------------------------------------------------------------------
// Kernel 1: fused input projections. Grid (M/128, H/16), block 256 (8 waves).
// Each wave owns one 16x16 output tile per gate; K tiled by 32.
// A-panel (128x32 bf16) staged via async-to-LDS; W tiles staged transposed.
// ---------------------------------------------------------------------------
__global__ void gru_xproj(const unsigned short* __restrict__ Xb,
                          const float* __restrict__ Wr,
                          const float* __restrict__ Wu,
                          const float* __restrict__ Wh,
                          const float* __restrict__ br,
                          const float* __restrict__ bu,
                          const float* __restrict__ bh,
                          float* __restrict__ XR,
                          float* __restrict__ XU,
                          float* __restrict__ XH) {
    // A tile 128x32 bf16 (row stride 40 to spread LDS banks, keeps 16B align)
    __shared__ unsigned short As[128 * 40];
    // W tiles stored transposed: [n=16][k=32], stride 40
    __shared__ unsigned short Bs[3][16 * 40];

    const int m0   = blockIdx.x * 128;
    const int n0   = blockIdx.y * 16;
    const int tid  = threadIdx.x;
    const int lane = tid & 31;
    const int wave = tid >> 5;

    v8f accR = {}, accU = {}, accH = {};

    const int lr = lane & 15;   // row within tile / col within tile
    const int kh = lane >> 4;   // K-half select

    const unsigned as_base = lds_lo32(&As[0]);

    for (int kk = 0; kk < I_; kk += 32) {
        __syncthreads();
        // ---- stage A: 128x32 bf16, pure async 16B copies (2 per thread) ----
        #pragma unroll
        for (int i = 0; i < 2; ++i) {
            int c  = tid + 256 * i;           // chunk 0..511
            int r  = c >> 2;                  // row 0..127
            int c8 = (c & 3) << 3;            // k sub-offset 0,8,16,24
            async_ld_b128(as_base + (unsigned)(r * 80 + (c & 3) * 16),
                          Xb + (size_t)(m0 + r) * I_ + kk + c8);
        }
        // ---- stage W tiles (32k x 16n fp32) transposed into LDS ----
        #pragma unroll
        for (int i = 0; i < 2; ++i) {
            int idx = tid + 256 * i;          // 0..511
            int n   = idx & 15;
            int k   = idx >> 4;               // 0..31
            size_t g = (size_t)(kk + k) * H_ + n0 + n;
            Bs[0][n * 40 + k] = f2bf(Wr[g]);
            Bs[1][n * 40 + k] = f2bf(Wu[g]);
            Bs[2][n * 40 + k] = f2bf(Wh[g]);
        }
        wait_async0();
        __syncthreads();

        // ---- fragments (contiguous k = 16*kh + s, same order in A and B) ----
        FragAB a;
        const unsigned short* ap = &As[(wave * 16 + lr) * 40 + kh * 16];
        a.q[0] = *(const uint4*)(ap);
        a.q[1] = *(const uint4*)(ap + 8);

        FragAB bR, bU, bH;
        const unsigned short* rp = &Bs[0][lr * 40 + kh * 16];
        bR.q[0] = *(const uint4*)(rp);  bR.q[1] = *(const uint4*)(rp + 8);
        const unsigned short* up = &Bs[1][lr * 40 + kh * 16];
        bU.q[0] = *(const uint4*)(up);  bU.q[1] = *(const uint4*)(up + 8);
        const unsigned short* hp = &Bs[2][lr * 40 + kh * 16];
        bH.q[0] = *(const uint4*)(hp);  bH.q[1] = *(const uint4*)(hp + 8);

        accR = __builtin_amdgcn_wmma_f32_16x16x32_bf16(
            false, a.v, false, bR.v, (short)0, accR, false, false);
        accU = __builtin_amdgcn_wmma_f32_16x16x32_bf16(
            false, a.v, false, bU.v, (short)0, accU, false, false);
        accH = __builtin_amdgcn_wmma_f32_16x16x32_bf16(
            false, a.v, false, bH.v, (short)0, accH, false, false);
    }

    // ---- epilogue: add bias, fp32 store (C layout: M=j / j+8, N=lane%16) ----
    const int n     = n0 + lr;
    const int rbase = m0 + wave * 16 + (kh ? 8 : 0);
    const float biasR = br[n], biasU = bu[n], biasH = bh[n];
    #pragma unroll
    for (int j = 0; j < 8; ++j) {
        size_t off = (size_t)(rbase + j) * H_ + n;
        XR[off] = accR[j] + biasR;
        XU[off] = accU[j] + biasU;
        XH[off] = accH[j] + biasH;
    }
}

// ---------------------------------------------------------------------------
// Global barrier across NWG workgroups (release add + acquire spin).
// goal is monotonically increasing -> no reset race.
// ---------------------------------------------------------------------------
#define NWG 32

__device__ __forceinline__ void global_barrier(unsigned int* cnt,
                                               unsigned int goal) {
    __syncthreads();
    if (threadIdx.x == 0) {
        __hip_atomic_fetch_add(cnt, 1u, __ATOMIC_RELEASE,
                               __HIP_MEMORY_SCOPE_AGENT);
        while (__hip_atomic_load(cnt, __ATOMIC_ACQUIRE,
                                 __HIP_MEMORY_SCOPE_AGENT) < goal) {
            __builtin_amdgcn_s_sleep(1);
        }
    }
    __syncthreads();
}

// ---------------------------------------------------------------------------
// Kernel 2: persistent recurrent scan.
// 32 WGs x 128 threads (4 waves). WG g owns H-columns [16g, 16g+16).
// Recurrent weight slices live in LDS (bf16, transposed, stride 520) for the
// whole scan. Two WMMA phases + two global barriers per timestep.
// ---------------------------------------------------------------------------
__global__ void gru_scan(const float* __restrict__ s0,
                         const float* __restrict__ Whr,
                         const float* __restrict__ Whu,
                         const float* __restrict__ Whh,
                         const float* __restrict__ XR,
                         const float* __restrict__ XU,
                         const float* __restrict__ XH,
                         unsigned short* __restrict__ state_bf,
                         unsigned short* __restrict__ rs_bf,
                         unsigned int* __restrict__ counter,
                         float* __restrict__ out) {
    // 3 x [16 cols][512 k] bf16, stride 520 (16B aligned, bank-spread): ~50KB
    __shared__ unsigned short Wt[3][16 * 520];

    const int wg   = blockIdx.x;      // 0..31
    const int n0   = wg * 16;
    const int tid  = threadIdx.x;
    const int lane = tid & 31;
    const int wave = tid >> 5;        // 0..3 -> m-tile

    // ---- stage recurrent weight slices once (fp32 -> bf16, transposed) ----
    for (int i = 0; i < 64; ++i) {
        int idx = tid + 128 * i;      // 0..8191
        int nl  = idx & 15;
        int k   = idx >> 4;           // 0..511
        size_t g = (size_t)k * H_ + n0 + nl;
        Wt[0][nl * 520 + k] = f2bf(Whr[g]);
        Wt[1][nl * 520 + k] = f2bf(Whu[g]);
        Wt[2][nl * 520 + k] = f2bf(Whh[g]);
    }
    __syncthreads();

    const int lr    = lane & 15;
    const int kh    = lane >> 4;
    const int mrow  = wave * 16;            // tile row base (B=64 -> 4 tiles)
    const int cn    = n0 + lr;              // this lane's output column
    const int rbase = mrow + (kh ? 8 : 0);  // C-layout row base

    unsigned int bar = 0;

    for (int t = 0; t < T_; ++t) {
        const float* prev =
            (t == 0) ? s0 : (out + (size_t)(t - 1) * B_ * H_);
        const size_t tb = (size_t)t * B_ * H_;

        // ---------- phase A: r,u GEMMs (state_bf @ Whr/Whu slice) ----------
        v8f aR = {}, aU = {};
        #pragma unroll
        for (int kk = 0; kk < H_; kk += 32) {
            FragAB a, bR, bU;
            const unsigned short* sp =
                state_bf + (size_t)(mrow + lr) * H_ + kk + kh * 16;
            a.q[0] = *(const uint4*)(sp);
            a.q[1] = *(const uint4*)(sp + 8);
            const unsigned short* rp = &Wt[0][lr * 520 + kk + kh * 16];
            bR.q[0] = *(const uint4*)(rp); bR.q[1] = *(const uint4*)(rp + 8);
            const unsigned short* up = &Wt[1][lr * 520 + kk + kh * 16];
            bU.q[0] = *(const uint4*)(up); bU.q[1] = *(const uint4*)(up + 8);
            aR = __builtin_amdgcn_wmma_f32_16x16x32_bf16(
                false, a.v, false, bR.v, (short)0, aR, false, false);
            aU = __builtin_amdgcn_wmma_f32_16x16x32_bf16(
                false, a.v, false, bU.v, (short)0, aU, false, false);
        }

        // elementwise: r = sigmoid(xr + .), u = sigmoid(xu + .), rs = r*state
        float uval[8];
        #pragma unroll
        for (int j = 0; j < 8; ++j) {
            size_t off = (size_t)(rbase + j) * H_ + cn;
            float s = prev[off];
            float r = sigmoidf_(aR[j] + XR[tb + off]);
            float u = sigmoidf_(aU[j] + XU[tb + off]);
            uval[j] = u;
            rs_bf[off] = f2bf(r * s);
            // warm next timestep's gate operands (global_prefetch_b8)
            if (t + 1 < T_) {
                __builtin_prefetch(XR + tb + (size_t)B_ * H_ + off, 0, 1);
                __builtin_prefetch(XU + tb + (size_t)B_ * H_ + off, 0, 1);
            }
        }

        global_barrier(counter, (++bar) * NWG);

        // ---------- phase B: c GEMM ((r*state) @ Whh slice) ----------
        v8f aC = {};
        #pragma unroll
        for (int kk = 0; kk < H_; kk += 32) {
            FragAB a, bH;
            const unsigned short* sp =
                rs_bf + (size_t)(mrow + lr) * H_ + kk + kh * 16;
            a.q[0] = *(const uint4*)(sp);
            a.q[1] = *(const uint4*)(sp + 8);
            const unsigned short* hp = &Wt[2][lr * 520 + kk + kh * 16];
            bH.q[0] = *(const uint4*)(hp); bH.q[1] = *(const uint4*)(hp + 8);
            aC = __builtin_amdgcn_wmma_f32_16x16x32_bf16(
                false, a.v, false, bH.v, (short)0, aC, false, false);
        }

        // elementwise: c = tanh(xh + .), h = u*s + (1-u)*c
        #pragma unroll
        for (int j = 0; j < 8; ++j) {
            size_t off = (size_t)(rbase + j) * H_ + cn;
            float s = prev[off];
            float c = tanhf(aC[j] + XH[tb + off]);
            float h = uval[j] * s + (1.0f - uval[j]) * c;
            out[tb + off] = h;                        // states[t]
            state_bf[off] = f2bf(h);                  // next step GEMM input
            if (t == T_ - 1)
                out[(size_t)T_ * B_ * H_ + off] = h;  // final_state
            if (t + 1 < T_)
                __builtin_prefetch(XH + tb + (size_t)B_ * H_ + off, 0, 1);
        }

        global_barrier(counter, (++bar) * NWG);
    }
}

// ---------------------------------------------------------------------------
extern "C" void kernel_launch(void* const* d_in, const int* in_sizes, int n_in,
                              void* d_out, int out_size, void* d_ws,
                              size_t ws_size, hipStream_t stream) {
    const float* X   = (const float*)d_in[0];
    const float* S0  = (const float*)d_in[1];
    const float* Wxr = (const float*)d_in[2];
    const float* Whr = (const float*)d_in[3];
    const float* br  = (const float*)d_in[4];
    const float* Wxu = (const float*)d_in[5];
    const float* Whu = (const float*)d_in[6];
    const float* bu  = (const float*)d_in[7];
    const float* Wxh = (const float*)d_in[8];
    const float* Whh = (const float*)d_in[9];
    const float* bh  = (const float*)d_in[10];
    float* out = (float*)d_out;

    const size_t TBH = (size_t)T_ * B_ * H_;
    float* XR = (float*)d_ws;
    float* XU = XR + TBH;
    float* XH = XU + TBH;
    unsigned short* Xb       = (unsigned short*)(XH + TBH);   // bf16 X copy
    unsigned short* state_bf = Xb + (size_t)T_ * B_ * I_;
    unsigned short* rs_bf    = state_bf + (size_t)B_ * H_;
    unsigned int*   counter  = (unsigned int*)(rs_bf + (size_t)B_ * H_);

    gru_init<<<dim3((B_ * H_ + 255) / 256), 256, 0, stream>>>(
        S0, state_bf, counter);

    x_to_bf16<<<dim3(((size_t)T_ * B_ * I_) / (256 * 8)), 256, 0, stream>>>(
        X, Xb);

    gru_xproj<<<dim3((T_ * B_) / 128, H_ / 16), 256, 0, stream>>>(
        Xb, Wxr, Wxu, Wxh, br, bu, bh, XR, XU, XH);

    gru_scan<<<dim3(NWG), 128, 0, stream>>>(
        S0, Whr, Whu, Whh, XR, XU, XH, state_bf, rs_bf, counter, out);
}
